// FusedMoE_39831526703663
// MI455X (gfx1250) — compile-verified, matrix operand order checked by reading, not run on hardware
//
#include <hip/hip_runtime.h>
#include <hip/hip_bf16.h>
#include <math.h>

#define T_TOKENS 4096
#define HID      2048
#define INTER    2048
#define NEXP     8

// Use CDNA5 async global->LDS copies (ASYNCcnt) for conversion-free A tiles.
#define MOE_USE_ASYNC 1

typedef _Float16 h16;
typedef __attribute__((ext_vector_type(16))) _Float16 v16h;
typedef __attribute__((ext_vector_type(8)))  float    v8f;

// ---------------------------------------------------------------------------
// Async 16B global->LDS copy (gfx1250 GLOBAL_LOAD_ASYNC_TO_LDS_B128).
// VDST carries the per-lane LDS byte address; tracked with ASYNCcnt.
// ---------------------------------------------------------------------------
__device__ inline void async_copy_b128(const void* gsrc, void* lds_dst) {
#if MOE_USE_ASYNC
  unsigned lds_off = (unsigned)(unsigned long long)lds_dst;  // LDS offset = addr[31:0]
  unsigned long long ga = (unsigned long long)gsrc;
  asm volatile("global_load_async_to_lds_b128 %0, %1, off"
               :: "v"(lds_off), "v"(ga)
               : "memory");
#else
  const unsigned long long* s = (const unsigned long long*)gsrc;
  unsigned long long q0 = s[0], q1 = s[1];
  unsigned long long* d = (unsigned long long*)lds_dst;
  d[0] = q0; d[1] = q1;
#endif
}

__device__ inline void async_wait() {
#if MOE_USE_ASYNC
  asm volatile("s_wait_asynccnt 0x0" ::: "memory");
#endif
}

// ---------------------------------------------------------------------------
// Fragment load from LDS for V_WMMA_F32_16X16X32_F16 (wave32).
// Lane L holds row (L&15); its 16 halves are K = (j/8)*16 + (L>=16)*8 + (j%8).
// Row stride 40 halves (80B) keeps these four 8B reads aligned (compiler
// merges them into ds_load_b128 pairs).
// ---------------------------------------------------------------------------
__device__ inline v16h frag_ld(const h16* rowbase, int hi) {
  const unsigned long long* p = (const unsigned long long*)rowbase;
  union { v16h v; unsigned long long q[4]; } u;
  int o = hi ? 2 : 0;
  u.q[0] = p[o + 0];
  u.q[1] = p[o + 1];
  u.q[2] = p[o + 4];
  u.q[3] = p[o + 5];
  return u.v;
}

// ---------------------------------------------------------------------------
// 0) x fp32 -> f16 once (x is reused ~3x: router + two experts per token).
// ---------------------------------------------------------------------------
__global__ void __launch_bounds__(256)
moe_cvt_x(const float* __restrict__ x, h16* __restrict__ x16) {
  size_t i = ((size_t)blockIdx.x * blockDim.x + threadIdx.x) * 8;
  float4 a = *(const float4*)(x + i);
  float4 b = *(const float4*)(x + i + 4);
  union { h16 h[8]; uint4 u; } r;
  r.h[0] = (h16)a.x; r.h[1] = (h16)a.y; r.h[2] = (h16)a.z; r.h[3] = (h16)a.w;
  r.h[4] = (h16)b.x; r.h[5] = (h16)b.y; r.h[6] = (h16)b.z; r.h[7] = (h16)b.w;
  *(uint4*)(x16 + i) = r.u;
}

// ---------------------------------------------------------------------------
// 1) Router: logits = x @ router_w, top-2, softmax over selected logits.
// ---------------------------------------------------------------------------
__global__ void __launch_bounds__(128)
moe_router(const float* __restrict__ x, const float* __restrict__ rw,
           int* __restrict__ tidx, float* __restrict__ tscale) {
  int wave = threadIdx.x >> 5;
  int lane = threadIdx.x & 31;
  int t = blockIdx.x * 4 + wave;
  if (t >= T_TOKENS) return;

  float acc[NEXP];
#pragma unroll
  for (int e = 0; e < NEXP; ++e) acc[e] = 0.f;

  const float* xr = x + (size_t)t * HID;
  for (int h = lane; h < HID; h += 32) {
    float xv = xr[h];
    const float* w = rw + (size_t)h * NEXP;
#pragma unroll
    for (int e = 0; e < NEXP; ++e) acc[e] += xv * w[e];
  }
#pragma unroll
  for (int e = 0; e < NEXP; ++e) {
#pragma unroll
    for (int s = 16; s > 0; s >>= 1) acc[e] += __shfl_xor(acc[e], s, 32);
  }
  if (lane == 0) {
    int i0 = 0; float v0 = acc[0];
#pragma unroll
    for (int e = 1; e < NEXP; ++e) if (acc[e] > v0) { v0 = acc[e]; i0 = e; }
    int i1 = (i0 == 0) ? 1 : 0; float v1 = acc[i1];
#pragma unroll
    for (int e = 0; e < NEXP; ++e)
      if (e != i0 && acc[e] > v1) { v1 = acc[e]; i1 = e; }
    float e1 = __expf(v1 - v0);          // v0 >= v1, stable softmax
    float inv = 1.f / (1.f + e1);
    tidx[t * 2 + 0] = i0;  tidx[t * 2 + 1] = i1;
    tscale[t * 2 + 0] = inv;  tscale[t * 2 + 1] = e1 * inv;
  }
}

// ---------------------------------------------------------------------------
// 2) Build packed per-expert token lists.
// ---------------------------------------------------------------------------
__global__ void moe_count(const int* __restrict__ tidx, int* __restrict__ counts) {
  int t = blockIdx.x * blockDim.x + threadIdx.x;
  if (t >= T_TOKENS) return;
  atomicAdd(&counts[tidx[t * 2 + 0]], 1);
  atomicAdd(&counts[tidx[t * 2 + 1]], 1);
}

__global__ void moe_scan(const int* __restrict__ counts, int* __restrict__ offs) {
  if (threadIdx.x == 0 && blockIdx.x == 0) {
    int s = 0;
    for (int e = 0; e < NEXP; ++e) { offs[e] = s; s += counts[e]; }
    offs[NEXP] = s;
  }
}

__global__ void moe_fill(const int* __restrict__ tidx, const float* __restrict__ tscale,
                         const int* __restrict__ offs, int* __restrict__ cursors,
                         int* __restrict__ perm, float* __restrict__ pscale) {
  int t = blockIdx.x * blockDim.x + threadIdx.x;
  if (t >= T_TOKENS) return;
#pragma unroll
  for (int k = 0; k < 2; ++k) {
    int e = tidx[t * 2 + k];
    int slot = atomicAdd(&cursors[e], 1);
    perm[offs[e] + slot]   = t;
    pscale[offs[e] + slot] = tscale[t * 2 + k];
  }
}

// ---------------------------------------------------------------------------
// 3) Gate/Up grouped GEMM + swiglu + token-scale, f16 output to hbuf.
//    A tile (gathered f16 x rows) streamed with async global->LDS copies.
// ---------------------------------------------------------------------------
__global__ void __launch_bounds__(256)
moe_gateup(const h16* __restrict__ x16,
           const float* __restrict__ w1,   // [E][H][I]
           const float* __restrict__ w3,   // [E][H][I]
           const int* __restrict__ perm, const float* __restrict__ pscale,
           const int* __restrict__ offs, h16* __restrict__ hbuf) {
  int e   = blockIdx.z;
  int off = offs[e];
  int cnt = offs[e + 1] - off;
  int mtile = blockIdx.y;
  if (mtile * 64 >= cnt) return;
  int n0 = blockIdx.x * 128;

  const float* W1 = w1 + (size_t)e * HID * INTER;
  const float* W3 = w3 + (size_t)e * HID * INTER;

  int tid  = threadIdx.x;
  int lane = tid & 31;
  int wave = tid >> 5;
  int wm = wave & 1;      // 0..1 -> 32-row half
  int wn = wave >> 1;     // 0..3 -> 32-col quarter

  __shared__ __align__(16) h16 As [64][40];
  __shared__ __align__(16) h16 B1s[128][40];   // stored transposed: [n][k]
  __shared__ __align__(16) h16 B3s[128][40];

  // A loader mapping: one b128 (8 halves) per thread
  int am  = tid >> 2;            // 0..63 token row in tile
  int akq = (tid & 3) * 8;       // k sub-offset (halves)
  int arow = mtile * 64 + am;
  int ar   = (arow < cnt) ? arow : cnt - 1;
  const h16* xrow = x16 + (size_t)perm[off + ar] * HID;

  // B loader mapping: 16 cols per thread per matrix
  int br  = tid >> 3;            // 0..31 k row
  int bc0 = (tid & 7) * 16;      // col sub-offset

  v8f zero = {0.f, 0.f, 0.f, 0.f, 0.f, 0.f, 0.f, 0.f};
  v8f accG[2][2], accU[2][2];
#pragma unroll
  for (int i = 0; i < 2; ++i)
#pragma unroll
    for (int j = 0; j < 2; ++j) { accG[i][j] = zero; accU[i][j] = zero; }

  for (int k0 = 0; k0 < HID; k0 += 32) {
    __syncthreads();
    // A tile: async byte copy (f16 already), ASYNCcnt-tracked
    async_copy_b128(xrow + k0 + akq, &As[am][akq]);
    {   // B tiles: w1/w3 rows k0..k0+31, cols n0..n0+127, cvt + transpose
      const float* p1 = W1 + (size_t)(k0 + br) * INTER + n0 + bc0;
      const float* p3 = W3 + (size_t)(k0 + br) * INTER + n0 + bc0;
#pragma unroll
      for (int j = 0; j < 16; j += 4) {
        float4 g = *(const float4*)(p1 + j);
        float4 u = *(const float4*)(p3 + j);
        B1s[bc0 + j + 0][br] = (h16)g.x; B1s[bc0 + j + 1][br] = (h16)g.y;
        B1s[bc0 + j + 2][br] = (h16)g.z; B1s[bc0 + j + 3][br] = (h16)g.w;
        B3s[bc0 + j + 0][br] = (h16)u.x; B3s[bc0 + j + 1][br] = (h16)u.y;
        B3s[bc0 + j + 2][br] = (h16)u.z; B3s[bc0 + j + 3][br] = (h16)u.w;
      }
      if (k0 + 32 < HID) {       // next K-step weight tile -> global_prefetch_b8
        __builtin_prefetch(p1 + (size_t)32 * INTER, 0, 1);
        __builtin_prefetch(p3 + (size_t)32 * INTER, 0, 1);
      }
    }
    async_wait();                // drain this wave's async LDS writes
    __syncthreads();

    int hi = lane >> 4;
    int r16 = lane & 15;
    v16h b1f[2], b3f[2];
#pragma unroll
    for (int ni = 0; ni < 2; ++ni) {
      b1f[ni] = frag_ld(&B1s[wn * 32 + ni * 16 + r16][0], hi);
      b3f[ni] = frag_ld(&B3s[wn * 32 + ni * 16 + r16][0], hi);
    }
#pragma unroll
    for (int mi = 0; mi < 2; ++mi) {
      v16h a = frag_ld(&As[wm * 32 + mi * 16 + r16][0], hi);
#pragma unroll
      for (int ni = 0; ni < 2; ++ni) {
        accG[mi][ni] = __builtin_amdgcn_wmma_f32_16x16x32_f16(
            false, a, false, b1f[ni], (short)0, accG[mi][ni], false, false);
        accU[mi][ni] = __builtin_amdgcn_wmma_f32_16x16x32_f16(
            false, a, false, b3f[ni], (short)0, accU[mi][ni], false, false);
      }
    }
  }

  // Epilogue: h = silu(gate) * up * token_scale  -> hbuf (f16)
  int r16 = lane & 15;
  int mhalf = 8 * (lane >> 4);
#pragma unroll
  for (int mi = 0; mi < 2; ++mi) {
#pragma unroll
    for (int r = 0; r < 8; ++r) {
      int m = mtile * 64 + wm * 32 + mi * 16 + r + mhalf;
      if (m < cnt) {
        float sc = pscale[off + m];
#pragma unroll
        for (int ni = 0; ni < 2; ++ni) {
          float g = accG[mi][ni][r];
          float u = accU[mi][ni][r];
          float h = (g / (1.f + __expf(-g))) * u * sc;
          int col = n0 + wn * 32 + ni * 16 + r16;
          hbuf[(size_t)(off + m) * INTER + col] = (h16)h;
        }
      }
    }
  }
}

// ---------------------------------------------------------------------------
// 4) Down projection: out[token] += hbuf_row @ w2[e]. A tile async-copied.
//    Combine via float atomics (each token has exactly 2 contributions,
//    2-operand fp add is commutative -> deterministic from zeroed out).
// ---------------------------------------------------------------------------
__global__ void __launch_bounds__(256)
moe_down(const h16* __restrict__ hbuf, const float* __restrict__ w2, // [E][I][H]
         const int* __restrict__ perm, const int* __restrict__ offs,
         float* __restrict__ out) {
  int e   = blockIdx.z;
  int off = offs[e];
  int cnt = offs[e + 1] - off;
  int mtile = blockIdx.y;
  if (mtile * 64 >= cnt) return;
  int n0 = blockIdx.x * 128;

  const float* W2 = w2 + (size_t)e * INTER * HID;

  int tid  = threadIdx.x;
  int lane = tid & 31;
  int wave = tid >> 5;
  int wm = wave & 1;
  int wn = wave >> 1;

  __shared__ __align__(16) h16 As[64][40];
  __shared__ __align__(16) h16 Bs[128][40];

  int am  = tid >> 2;
  int akq = (tid & 3) * 8;
  int arow = mtile * 64 + am;
  int ar   = (arow < cnt) ? arow : cnt - 1;
  const h16* hrow = hbuf + (size_t)(off + ar) * INTER;

  int br  = tid >> 3;
  int bc0 = (tid & 7) * 16;

  v8f zero = {0.f, 0.f, 0.f, 0.f, 0.f, 0.f, 0.f, 0.f};
  v8f acc[2][2];
#pragma unroll
  for (int i = 0; i < 2; ++i)
#pragma unroll
    for (int j = 0; j < 2; ++j) acc[i][j] = zero;

  for (int k0 = 0; k0 < INTER; k0 += 32) {
    __syncthreads();
    // A tile: async byte copy of packed f16 activations
    async_copy_b128(hrow + k0 + akq, &As[am][akq]);
    {   // B tile: w2 rows k0..k0+31, cols n0..n0+127, cvt + transpose
      const float* p2 = W2 + (size_t)(k0 + br) * HID + n0 + bc0;
#pragma unroll
      for (int j = 0; j < 16; j += 4) {
        float4 g = *(const float4*)(p2 + j);
        Bs[bc0 + j + 0][br] = (h16)g.x; Bs[bc0 + j + 1][br] = (h16)g.y;
        Bs[bc0 + j + 2][br] = (h16)g.z; Bs[bc0 + j + 3][br] = (h16)g.w;
      }
      if (k0 + 32 < INTER)
        __builtin_prefetch(p2 + (size_t)32 * HID, 0, 1);
    }
    async_wait();
    __syncthreads();

    int hi = lane >> 4;
    int r16 = lane & 15;
    v16h bf[2];
#pragma unroll
    for (int ni = 0; ni < 2; ++ni)
      bf[ni] = frag_ld(&Bs[wn * 32 + ni * 16 + r16][0], hi);
#pragma unroll
    for (int mi = 0; mi < 2; ++mi) {
      v16h a = frag_ld(&As[wm * 32 + mi * 16 + r16][0], hi);
#pragma unroll
      for (int ni = 0; ni < 2; ++ni)
        acc[mi][ni] = __builtin_amdgcn_wmma_f32_16x16x32_f16(
            false, a, false, bf[ni], (short)0, acc[mi][ni], false, false);
    }
  }

  int r16 = lane & 15;
  int mhalf = 8 * (lane >> 4);
#pragma unroll
  for (int mi = 0; mi < 2; ++mi) {
#pragma unroll
    for (int r = 0; r < 8; ++r) {
      int m = mtile * 64 + wm * 32 + mi * 16 + r + mhalf;
      if (m < cnt) {
        int t = perm[off + m];
#pragma unroll
        for (int ni = 0; ni < 2; ++ni) {
          int col = n0 + wn * 32 + ni * 16 + r16;
          atomicAdd(&out[(size_t)t * HID + col], acc[mi][ni][r]);
        }
      }
    }
  }
}

// ---------------------------------------------------------------------------
// Host-side launch
// ---------------------------------------------------------------------------
extern "C" void kernel_launch(void* const* d_in, const int* in_sizes, int n_in,
                              void* d_out, int out_size, void* d_ws, size_t ws_size,
                              hipStream_t stream) {
  const float* x  = (const float*)d_in[0];
  const float* rw = (const float*)d_in[1];
  const float* w1 = (const float*)d_in[2];
  const float* w3 = (const float*)d_in[3];
  const float* w2 = (const float*)d_in[4];
  float* out = (float*)d_out;

  // Workspace carve-up (256B aligned slices)
  char* ws = (char*)d_ws;
  size_t o = 0;
  auto take = [&](size_t bytes) {
    char* p = ws + o;
    o += (bytes + 255) & ~(size_t)255;
    return p;
  };
  h16*   hbuf   = (h16*)  take((size_t)2 * T_TOKENS * INTER * sizeof(h16)); // 32 MiB
  h16*   x16    = (h16*)  take((size_t)T_TOKENS * HID * sizeof(h16));       // 16 MiB
  int*   tidx   = (int*)  take((size_t)T_TOKENS * 2 * sizeof(int));
  float* tscale = (float*)take((size_t)T_TOKENS * 2 * sizeof(float));
  int*   perm   = (int*)  take((size_t)T_TOKENS * 2 * sizeof(int));
  float* pscale = (float*)take((size_t)T_TOKENS * 2 * sizeof(float));
  int*   counts = (int*)  take((size_t)NEXP * sizeof(int));
  int*   cursors= (int*)  take((size_t)NEXP * sizeof(int));
  int*   offs   = (int*)  take((size_t)(NEXP + 1) * sizeof(int));
  (void)in_sizes; (void)n_in; (void)ws_size;

  hipMemsetAsync(counts,  0, NEXP * sizeof(int), stream);
  hipMemsetAsync(cursors, 0, NEXP * sizeof(int), stream);
  hipMemsetAsync(d_out,   0, (size_t)out_size * sizeof(float), stream);

  moe_cvt_x <<<(T_TOKENS * HID) / (256 * 8), 256, 0, stream>>>(x, x16);
  moe_router<<<(T_TOKENS + 3) / 4, 128, 0, stream>>>(x, rw, tidx, tscale);
  moe_count <<<T_TOKENS / 256, 256, 0, stream>>>(tidx, counts);
  moe_scan  <<<1, 32, 0, stream>>>(counts, offs);
  moe_fill  <<<T_TOKENS / 256, 256, 0, stream>>>(tidx, tscale, offs, cursors, perm, pscale);

  dim3 g2(INTER / 128, (T_TOKENS + 63) / 64, NEXP);
  moe_gateup<<<g2, 256, 0, stream>>>(x16, w1, w3, perm, pscale, offs, hbuf);

  dim3 g3(HID / 128, (T_TOKENS + 63) / 64, NEXP);
  moe_down<<<g3, 256, 0, stream>>>(hbuf, w2, perm, offs, out);
}
